// RelationalEmbedding_33767032881919
// MI455X (gfx1250) — matrix operand order repaired and checked by reading, not compiled, not run on hardware
//
#include <hip/hip_runtime.h>

// ---------------------------------------------------------------------------
// RelationalEmbedding forward for MI455X (gfx1250), wave32 + WMMA bf16 + TDM.
//
//   q = O0 Ww^T + Wb ; k = O0 Uw^T + Ub ; v = O0 Hw^T + Hb
//   S = q k^T ; P = softmax_rows(S) ; ctx = P v
//   O1 = O0 + ctx fc0w^T + fc0b ; O2 = O1 fc1w^T + fc1b
//
// GEMMs: v_wmma_f32_16x16x32_bf16, f32 accumulate. Block tile 256x128, BK=32,
// 8 waves each computing 64x64 (16 wmma / K-step, 1:1 wmma:ds_load_b128).
// Tile staging: Tensor Data Mover (tensor_load_to_lds) with double-buffered
// LDS, gated by s_wait_tensorcnt + barrier; falls back to global->LDS staging
// if the builtin is unavailable.
// ---------------------------------------------------------------------------

typedef __bf16 bf16;
typedef __attribute__((ext_vector_type(16))) __bf16 v16bf;
typedef __attribute__((ext_vector_type(8)))  __bf16 bf16x8;
typedef __attribute__((ext_vector_type(8)))  float  v8f;
typedef __attribute__((ext_vector_type(4)))  unsigned short ushort4v;

#if __has_builtin(__builtin_amdgcn_tensor_load_to_lds) && \
    __has_builtin(__builtin_amdgcn_s_wait_tensorcnt)
#define USE_TDM 1
#else
#define USE_TDM 0
#endif

__device__ __forceinline__ unsigned short f2bfu(float x) {
  union { float f; unsigned u; } c; c.f = x;
  unsigned r = c.u + 0x7FFFu + ((c.u >> 16) & 1u);   // round-to-nearest-even
  return (unsigned short)(r >> 16);
}
__device__ __forceinline__ bf16 f2bf(float x) {
  union { unsigned short s; bf16 b; } o; o.s = f2bfu(x);
  return o.b;
}

// ---------------------------------------------------------------------------
// f32 -> bf16 conversion (one-time, grid-stride)
// ---------------------------------------------------------------------------
__global__ void __launch_bounds__(256)
cvt_f32_to_bf16(const float* __restrict__ src, bf16* __restrict__ dst, long n) {
  long i = (long)blockIdx.x * blockDim.x + threadIdx.x;
  long stride = (long)gridDim.x * blockDim.x;
  for (; i < n; i += stride) dst[i] = f2bf(src[i]);
}

#if USE_TDM
typedef __attribute__((ext_vector_type(4))) unsigned int uint32x4;
typedef __attribute__((ext_vector_type(8))) int          int32x8;
typedef __attribute__((ext_vector_type(4))) int          int32x4;

// Issue one TDM tile load: tile_d1 rows x tile_d0 elems (2B each) from a
// row-major tensor with row stride `stride_el`, into LDS at byte offset
// lds_off, inserting 16B of LDS padding after every 64B row (matches the
// [rows][BK+8] LDS layout used by the compute side).
// D# packing per CDNA5 ISA §8 (group0: count/lds/global/type, group1:
// data_size=2B, pad_interval=3 (16 DW), pad_amount=3 (4 DW), dims/strides).
__device__ __forceinline__ void
tdm_load_tile(const bf16* gtile, unsigned lds_off, unsigned tile_d0,
              unsigned tile_d1, unsigned stride_el) {
  unsigned long long ga = (unsigned long long)(uintptr_t)gtile;
  uint32x4 g0;
  g0.x = 1u;                                         // count=1, user desc
  g0.y = lds_off;                                    // LDS byte address
  g0.z = (unsigned)(ga & 0xFFFFFFFFu);               // global_addr[31:0]
  g0.w = (unsigned)((ga >> 32) & 0x01FFFFFFu) | (2u << 30);  // [56:32]|type=2
  int32x8 g1;
  g1[0] = (int)((1u << 16) |                         // data_size = 2 bytes
                (1u << 20) |                         // pad_enable
                (3u << 22) |                         // pad_interval: 16 DWORDs
                (3u << 25));                         // pad_amount: 4 DWORDs
  g1[1] = (int)((tile_d0 & 0xFFFFu) << 16);          // tensor_dim0[15:0]
  g1[2] = (int)(((tile_d0 >> 16) & 0xFFFFu) | ((tile_d1 & 0xFFFFu) << 16));
  g1[3] = (int)(((tile_d1 >> 16) & 0xFFFFu) | ((tile_d0 & 0xFFFFu) << 16));
  g1[4] = (int)(tile_d1 & 0xFFFFu);                  // tile_dim1 (tile_dim2=0)
  g1[5] = (int)stride_el;                            // tensor_dim0_stride[31:0]
  g1[6] = 0;                                         // stride[47:32], s1 lo
  g1[7] = 0;
  int32x4 z4 = {0, 0, 0, 0};
#if defined(__clang_major__) && __clang_major__ >= 23
  int32x8 z8 = {0, 0, 0, 0, 0, 0, 0, 0};
  __builtin_amdgcn_tensor_load_to_lds(g0, g1, z4, z4, z8, 0);
#else
  __builtin_amdgcn_tensor_load_to_lds(g0, g1, z4, z4, 0);
#endif
}

__device__ __forceinline__ unsigned lds_byte_off(const void* p) {
  // generic pointer to LDS: addr[31:0] is the LDS offset (aperture in [63:32])
  return (unsigned)(uintptr_t)p;
}
#endif  // USE_TDM

// ---------------------------------------------------------------------------
// C[M,N] = A[M,K] * B[N,K]^T (+bias[n]) (+resid[m,n])
// A,B bf16 K-contiguous. Optional f32 store (Cf), bf16 store (Ch).
// TRANS=true: store bf16 transposed Ch[n*ldc + m] (packed 8-wide along M).
// Block: 256 thr = 8 waves (4 in M x 2 in N). Block tile 256x128, BK=32.
// Per wave: 4x4 grid of 16x16 tiles -> 16 wmma per K-step.
// All dims here are multiples of the tile sizes -> no guards.
// ---------------------------------------------------------------------------
template <bool TRANS>
__global__ void __launch_bounds__(256)
gemm_bf16_wmma(const bf16* __restrict__ A, int lda,
               const bf16* __restrict__ B, int ldb,
               const float* __restrict__ bias,
               const float* __restrict__ resid,
               float* __restrict__ Cf,
               bf16*  __restrict__ Ch, int ldc,
               int M, int N, int K) {
  constexpr int BM = 256, BN = 128, BK = 32, LDP = BK + 8;  // +8: bank pad
#if USE_TDM
  __shared__ bf16 As[2][BM][LDP];
  __shared__ bf16 Bs[2][BN][LDP];
#else
  __shared__ bf16 As[1][BM][LDP];
  __shared__ bf16 Bs[1][BN][LDP];
#endif

  const int tid     = threadIdx.x;
  const int lane    = tid & 31;
  const int wave    = tid >> 5;
  const int waveM   = wave >> 1;     // 0..3
  const int waveN   = wave & 1;      // 0..1
  const int blockM  = blockIdx.y * BM;
  const int blockN  = blockIdx.x * BN;
  const int halfsel = lane >> 4;     // 0: lanes 0-15, 1: lanes 16-31
  const int l16     = lane & 15;

  v8f acc[4][4];
  const v8f zero = {0.f, 0.f, 0.f, 0.f, 0.f, 0.f, 0.f, 0.f};
#pragma unroll
  for (int mi = 0; mi < 4; ++mi)
#pragma unroll
    for (int ni = 0; ni < 4; ++ni) acc[mi][ni] = zero;

  const int nIter = K / BK;

#if USE_TDM
  // ---- TDM mainloop: double-buffered LDS, wave 0 drives the DMA ----------
  const bool issuer = (tid < 32);
  const bf16* Atile = A + (size_t)blockM * lda;
  const bf16* Btile = B + (size_t)blockN * ldb;
  if (issuer) {
    tdm_load_tile(Atile, lds_byte_off(&As[0][0][0]), BK, BM, (unsigned)lda);
    tdm_load_tile(Btile, lds_byte_off(&Bs[0][0][0]), BK, BN, (unsigned)ldb);
  }
  for (int it = 0; it < nIter; ++it) {
    const int cur = it & 1;
    if (issuer) {
      if (it + 1 < nIter) {
        const int kn = (it + 1) * BK;
        tdm_load_tile(Atile + kn, lds_byte_off(&As[cur ^ 1][0][0]), BK, BM,
                      (unsigned)lda);
        tdm_load_tile(Btile + kn, lds_byte_off(&Bs[cur ^ 1][0][0]), BK, BN,
                      (unsigned)ldb);
        __builtin_amdgcn_s_wait_tensorcnt(2);  // oldest pair (cur) done
      } else {
        __builtin_amdgcn_s_wait_tensorcnt(0);
      }
    }
    __syncthreads();   // cur buffer visible to all waves
#else
  // ---- fallback mainloop: staged global->LDS, single buffer --------------
  for (int it = 0; it < nIter; ++it) {
    const int cur = 0;
    const int k0 = it * BK;
    __syncthreads();   // previous iteration's LDS reads done
    // A tile 256x32: 4 x b128 per thread
#pragma unroll
    for (int t = 0; t < 4; ++t) {
      int ci = tid + t * 256;
      int row = ci >> 2, c = ci & 3;
      *(bf16x8*)&As[0][row][c * 8] =
          *(const bf16x8*)(A + (size_t)(blockM + row) * lda + k0 + c * 8);
    }
    // B tile 128x32: 2 x b128 per thread
#pragma unroll
    for (int t = 0; t < 2; ++t) {
      int ci = tid + t * 256;
      int row = ci >> 2, c = ci & 3;
      *(bf16x8*)&Bs[0][row][c * 8] =
          *(const bf16x8*)(B + (size_t)(blockN + row) * ldb + k0 + c * 8);
    }
    __syncthreads();
#endif

    // ---- fragments (ISA 7.12.2 layouts) ----
    union Frag { v16bf v; bf16x8 h[2]; } af[4], bfr[4];
#pragma unroll
    for (int mi = 0; mi < 4; ++mi) {
      int r = waveM * 64 + mi * 16 + l16;
      // A 16x32: lanes 0-15 hold K 0..7 & 16..23 ; lanes 16-31: 8..15 & 24..31
      af[mi].h[0] = *(const bf16x8*)&As[cur][r][halfsel * 8];
      af[mi].h[1] = *(const bf16x8*)&As[cur][r][16 + halfsel * 8];
    }
#pragma unroll
    for (int ni = 0; ni < 4; ++ni) {
      int r = waveN * 64 + ni * 16 + l16;
      // B 32x16: lanes 0-15 hold K 0..7 & 16..23 ; lanes 16-31: 8..15 & 24..31
      bfr[ni].h[0] = *(const bf16x8*)&Bs[cur][r][halfsel * 8];
      bfr[ni].h[1] = *(const bf16x8*)&Bs[cur][r][16 + halfsel * 8];
    }
#pragma unroll
    for (int mi = 0; mi < 4; ++mi)
#pragma unroll
      for (int ni = 0; ni < 4; ++ni)
        acc[mi][ni] = __builtin_amdgcn_wmma_f32_16x16x32_bf16(
            false, af[mi].v, false, bfr[ni].v, (short)0, acc[mi][ni],
            false, false);
    __syncthreads();   // all reads of cur done before it is refilled
  }

  // ---- epilogue. C layout: VGPR j, lanes 0-15: M=j,N=lane ; 16-31: M=j+8 --
#pragma unroll
  for (int mi = 0; mi < 4; ++mi) {
#pragma unroll
    for (int ni = 0; ni < 4; ++ni) {
      const int n  = blockN + waveN * 64 + ni * 16 + l16;
      const int m0 = blockM + waveM * 64 + mi * 16 + halfsel * 8;
      const float b = bias ? bias[n] : 0.0f;
      if constexpr (TRANS) {
        bf16x8 pk;
#pragma unroll
        for (int j = 0; j < 8; ++j) pk[j] = f2bf(acc[mi][ni][j] + b);
        *(bf16x8*)(Ch + (size_t)n * ldc + m0) = pk;   // 16B packed along M
      } else {
#pragma unroll
        for (int j = 0; j < 8; ++j) {
          const int m = m0 + j;
          float v = acc[mi][ni][j] + b;
          if (resid) v += resid[(size_t)m * ldc + n];
          if (Cf) Cf[(size_t)m * ldc + n] = v;
          if (Ch) Ch[(size_t)m * ldc + n] = f2bf(v);
        }
      }
    }
  }
}

// ---------------------------------------------------------------------------
// Row softmax, in place: reads 8192 f32, writes 8192 bf16 over the row start.
// One 256-thread block per row; whole row held in registers (32 f32/thread).
// ---------------------------------------------------------------------------
__global__ void __launch_bounds__(256)
softmax_row_inplace_bf16(float* __restrict__ S, int ncols) {
  const int row = blockIdx.x;
  const int tid = threadIdx.x;
  float4* r4 = (float4*)(S + (size_t)row * ncols);

  float4 v[8];
  float mx = -3.4e38f;
#pragma unroll
  for (int i = 0; i < 8; ++i) {
    v[i] = r4[tid + i * 256];
    mx = fmaxf(mx, fmaxf(fmaxf(v[i].x, v[i].y), fmaxf(v[i].z, v[i].w)));
  }
  __shared__ float red[256];
  red[tid] = mx;
  __syncthreads();
  for (int s = 128; s > 0; s >>= 1) {
    if (tid < s) red[tid] = fmaxf(red[tid], red[tid + s]);
    __syncthreads();
  }
  const float rowmax = red[0];
  __syncthreads();

  float sum = 0.f;
#pragma unroll
  for (int i = 0; i < 8; ++i) {
    v[i].x = __expf(v[i].x - rowmax);
    v[i].y = __expf(v[i].y - rowmax);
    v[i].z = __expf(v[i].z - rowmax);
    v[i].w = __expf(v[i].w - rowmax);
    sum += v[i].x + v[i].y + v[i].z + v[i].w;
  }
  red[tid] = sum;
  __syncthreads();
  for (int s = 128; s > 0; s >>= 1) {
    if (tid < s) red[tid] += red[tid + s];
    __syncthreads();
  }
  const float inv = 1.0f / red[0];

  ushort4v* out = (ushort4v*)(S + (size_t)row * ncols);  // bf16 over same row
#pragma unroll
  for (int i = 0; i < 8; ++i) {
    ushort4v q;
    q.x = f2bfu(v[i].x * inv);
    q.y = f2bfu(v[i].y * inv);
    q.z = f2bfu(v[i].z * inv);
    q.w = f2bfu(v[i].w * inv);
    out[tid + i * 256] = q;
  }
}

// ---------------------------------------------------------------------------
extern "C" void kernel_launch(void* const* d_in, const int* in_sizes, int n_in,
                              void* d_out, int out_size, void* d_ws,
                              size_t ws_size, hipStream_t stream) {
  (void)in_sizes; (void)n_in; (void)out_size; (void)ws_size;
  const int Nn = 8192, D = 2048, DH = 1024, DO = 1024;

  const float* O0   = (const float*)d_in[0];
  const float* Ww   = (const float*)d_in[1];
  const float* Wb   = (const float*)d_in[2];
  const float* Uw   = (const float*)d_in[3];
  const float* Ub   = (const float*)d_in[4];
  const float* Hw   = (const float*)d_in[5];
  const float* Hb   = (const float*)d_in[6];
  const float* fc0w = (const float*)d_in[7];
  const float* fc0b = (const float*)d_in[8];
  const float* fc1w = (const float*)d_in[9];
  const float* fc1b = (const float*)d_in[10];

  float* O1f = (float*)d_out;                       // [8192, 2048]
  float* O2f = (float*)d_out + (size_t)Nn * D;      // [8192, 1024]

  // ---- deterministic ws carve (~424 MB) ----
  char* w = (char*)d_ws;
  auto take = [&](size_t bytes) -> void* {
    void* p = (void*)w;
    w += (bytes + 255) & ~(size_t)255;
    return p;
  };
  bf16* O0h   = (bf16*)take((size_t)Nn * D  * 2);
  bf16* Wwh   = (bf16*)take((size_t)DH * D  * 2);
  bf16* Uwh   = (bf16*)take((size_t)DH * D  * 2);
  bf16* Hwh   = (bf16*)take((size_t)DH * D  * 2);
  bf16* fc0wh = (bf16*)take((size_t)D  * DH * 2);
  bf16* fc1wh = (bf16*)take((size_t)DO * D  * 2);
  bf16* Qh    = (bf16*)take((size_t)Nn * DH * 2);
  bf16* Kh    = (bf16*)take((size_t)Nn * DH * 2);
  bf16* Vth   = (bf16*)take((size_t)DH * Nn * 2);   // transposed V
  bf16* Ctxh  = (bf16*)take((size_t)Nn * DH * 2);
  bf16* O1h   = (bf16*)take((size_t)Nn * D  * 2);
  float* Sf   = (float*)take((size_t)Nn * Nn * 4);  // scores (bf16 P in place)

  // ---- 1. convert inputs/weights to bf16 ----
  cvt_f32_to_bf16<<<4096, 256, 0, stream>>>(O0,   O0h,   (long)Nn * D);
  cvt_f32_to_bf16<<<2048, 256, 0, stream>>>(Ww,   Wwh,   (long)DH * D);
  cvt_f32_to_bf16<<<2048, 256, 0, stream>>>(Uw,   Uwh,   (long)DH * D);
  cvt_f32_to_bf16<<<2048, 256, 0, stream>>>(Hw,   Hwh,   (long)DH * D);
  cvt_f32_to_bf16<<<2048, 256, 0, stream>>>(fc0w, fc0wh, (long)D  * DH);
  cvt_f32_to_bf16<<<2048, 256, 0, stream>>>(fc1w, fc1wh, (long)DO * D);

  const dim3 blk(256);
  // ---- 2. QKV projections (M=8192, N=1024, K=2048) ----
  gemm_bf16_wmma<false><<<dim3(DH / 128, Nn / 256), blk, 0, stream>>>(
      O0h, D, Wwh, D, Wb, nullptr, nullptr, Qh, DH, Nn, DH, D);
  gemm_bf16_wmma<false><<<dim3(DH / 128, Nn / 256), blk, 0, stream>>>(
      O0h, D, Uwh, D, Ub, nullptr, nullptr, Kh, DH, Nn, DH, D);
  gemm_bf16_wmma<true><<<dim3(DH / 128, Nn / 256), blk, 0, stream>>>(
      O0h, D, Hwh, D, Hb, nullptr, nullptr, Vth, Nn, Nn, DH, D);

  // ---- 3. scores S = q k^T (M=N=8192, K=1024), f32 out ----
  gemm_bf16_wmma<false><<<dim3(Nn / 128, Nn / 256), blk, 0, stream>>>(
      Qh, DH, Kh, DH, nullptr, nullptr, Sf, nullptr, Nn, Nn, Nn, DH);

  // ---- 4. softmax rows, P written bf16 in place (row stride 2*Nn bf16) ----
  softmax_row_inplace_bf16<<<Nn, blk, 0, stream>>>(Sf, Nn);

  // ---- 5. ctx = P v  (A = bf16 P, lda = 2*Nn; B = Vt, K = 8192) ----
  gemm_bf16_wmma<false><<<dim3(DH / 128, Nn / 256), blk, 0, stream>>>(
      (const bf16*)Sf, 2 * Nn, Vth, Nn, nullptr, nullptr, nullptr, Ctxh, DH,
      Nn, DH, Nn);

  // ---- 6. O1 = O0 + ctx fc0w^T + fc0b (f32 out + bf16 copy) ----
  gemm_bf16_wmma<false><<<dim3(D / 128, Nn / 256), blk, 0, stream>>>(
      Ctxh, DH, fc0wh, DH, fc0b, O0, O1f, O1h, D, Nn, D, DH);

  // ---- 7. O2 = O1 fc1w^T + fc1b ----
  gemm_bf16_wmma<false><<<dim3(DO / 128, Nn / 256), blk, 0, stream>>>(
      O1h, D, fc1wh, D, fc1b, nullptr, O2f, nullptr, DO, Nn, DO, D);
}
